// Net_89077621719646
// MI455X (gfx1250) — matrix operand order repaired and checked by reading, not compiled, not run on hardware
//
#include <hip/hip_runtime.h>
#include <stdint.h>

// ---------------------------------------------------------------------------
// Bidirectional LSTM + attention pooling + FC head for MI455X (gfx1250).
//   B=64, T=512, E=256, H=512, G=4H=2048, OH=1024
// Strategy:
//   1) Convert x / reversed-x / weights to bf16; fuse bih+bhh.
//   2) x_proj = x @ Wih^T + bias via v_wmma_f32_16x16x32_bf16, stored [T][B][G]
//      fp32 so each recurrence step reads a contiguous 512KB slab.
//   3) Recurrence: one persistent 1024-thread workgroup per direction.
//      32 waves; wave w owns hidden units [16w,16w+16): its 4 gate tiles
//      (i,f,g,o) x 4 M-tiles accumulate in VGPRs, cell state stays in VGPRs,
//      h double-buffered as bf16 in LDS (2*64*512*2 = 128KB), one barrier/step.
//      Whh (2MB bf16) streams from L2 every step.
//   4) Backward outputs pooled directly with reversed softmax weights
//      (t' = len-1-t change of variable) -- no un-reverse gather needed.
// ---------------------------------------------------------------------------

typedef __attribute__((ext_vector_type(16))) __bf16 v16bf;
typedef __attribute__((ext_vector_type(8)))  float  v8f;

#define T_SEQ 512
#define B_SZ  64
#define E_SZ  256
#define H_SZ  512
#define G_SZ  2048
#define OH_SZ 1024

union FragAB {          // 16 bf16 = 32 bytes = operand of one 16x16x32 WMMA
  v16bf v;
  uint4 q[2];
  unsigned short u[16];
};

__device__ __forceinline__ unsigned short f2bf(float f) {
  union { float f; uint32_t u; } x; x.f = f;
  uint32_t r = x.u + 0x7FFFu + ((x.u >> 16) & 1u);   // round-to-nearest-even
  return (unsigned short)(r >> 16);
}

__device__ __forceinline__ float sigmoidf_(float x) {
  return 1.f / (1.f + __expf(-x));
}

// ---------------------------------------------------------------- prep kernels
__global__ __launch_bounds__(256) void cvt_bf16_kernel(
    const float* __restrict__ src, unsigned short* __restrict__ dst, int n) {
  int i = blockIdx.x * 256 + threadIdx.x;
  if (i < n) dst[i] = f2bf(src[i]);
}

__global__ __launch_bounds__(256) void add_bias_kernel(
    const float* __restrict__ a, const float* __restrict__ b,
    float* __restrict__ o, int n) {
  int i = blockIdx.x * 256 + threadIdx.x;
  if (i < n) o[i] = a[i] + b[i];
}

// Xf[b*T+t][e] = bf16(x[b][t][e]);  Xb = per-batch time-reversed (zero-padded)
__global__ __launch_bounds__(256) void prep_x_kernel(
    const float* __restrict__ x, const int* __restrict__ lengths,
    unsigned short* __restrict__ Xf, unsigned short* __restrict__ Xb) {
  int i = blockIdx.x * 256 + threadIdx.x;
  if (i >= B_SZ * T_SEQ * E_SZ) return;
  int e  = i & (E_SZ - 1);
  int bt = i >> 8;
  int t  = bt & (T_SEQ - 1);
  int b  = bt >> 9;
  Xf[i] = f2bf(x[i]);
  int idx = lengths[b] - 1 - t;
  Xb[i] = (idx >= 0) ? f2bf(x[((size_t)(b * T_SEQ + idx) << 8) + e])
                     : (unsigned short)0;
}

// ------------------------------------------------------- x-projection GEMM
// C[M=BT x N=G] = X[BT x E] * W^T,  W row-major [G][E] (so W row n is B-col n).
// Block: 256 thr = 8 waves; block tile 64(M) x 256(N); wave tile 64 x 32.
__global__ __launch_bounds__(256) void xproj_gemm_kernel(
    const unsigned short* __restrict__ Xf, const unsigned short* __restrict__ Xb,
    const unsigned short* __restrict__ Wf, const unsigned short* __restrict__ Wb,
    const float* __restrict__ bias_f, const float* __restrict__ bias_b,
    float* __restrict__ Pf, float* __restrict__ Pb) {
  const unsigned short* X = blockIdx.z ? Xb : Xf;
  const unsigned short* W = blockIdx.z ? Wb : Wf;
  const float* bias       = blockIdx.z ? bias_b : bias_f;
  float* P                = blockIdx.z ? Pb : Pf;

  const int wave = threadIdx.x >> 5;
  const int lane = threadIdx.x & 31;
  const int lr   = lane & 15;          // row/col within tile
  const int hi   = lane >> 4;          // lane half selects K/M sub-block

  const int m_base = blockIdx.x * 64;
  const int n_wave = blockIdx.y * 256 + wave * 32;

  v8f zero = {0.f,0.f,0.f,0.f,0.f,0.f,0.f,0.f};
  v8f acc[4][2];
#pragma unroll
  for (int m = 0; m < 4; ++m)
#pragma unroll
    for (int nt = 0; nt < 2; ++nt) acc[m][nt] = zero;

  for (int ks = 0; ks < E_SZ / 32; ++ks) {
    const int k0 = ks * 32;
    FragAB a[4], bf[2];
#pragma unroll
    for (int m = 0; m < 4; ++m) {   // A 16x32 bf16: lanes<16 K{0..7,16..23}, hi K{8..15,24..31}
      const unsigned short* ap =
          X + (size_t)(m_base + m * 16 + lr) * E_SZ + k0 + hi * 8;
      a[m].q[0] = *(const uint4*)ap;
      a[m].q[1] = *(const uint4*)(ap + 16);
    }
#pragma unroll
    for (int nt = 0; nt < 2; ++nt) { // B 32x16 bf16: lanes<16 K{0..7,16..23}, hi K{8..15,24..31} of col lr
      const unsigned short* bp =
          W + (size_t)(n_wave + nt * 16 + lr) * E_SZ + k0 + hi * 8;
      bf[nt].q[0] = *(const uint4*)bp;
      bf[nt].q[1] = *(const uint4*)(bp + 16);
    }
#pragma unroll
    for (int m = 0; m < 4; ++m)
#pragma unroll
      for (int nt = 0; nt < 2; ++nt)
        acc[m][nt] = __builtin_amdgcn_wmma_f32_16x16x32_bf16(
            false, a[m].v, false, bf[nt].v, (short)0, acc[m][nt], false, false);
  }

  // C layout: vgpr r -> M = m_base + r + hi*8, N = n_base + lr.  Store [t][b][g].
#pragma unroll
  for (int m = 0; m < 4; ++m) {
    const int row0 = m_base + m * 16 + hi * 8;
#pragma unroll
    for (int nt = 0; nt < 2; ++nt) {
      const int col = n_wave + nt * 16 + lr;
      const float bv = bias[col];
#pragma unroll
      for (int r = 0; r < 8; ++r) {
        const int row = row0 + r;
        const int b_ = row >> 9;
        const int t_ = row & (T_SEQ - 1);
        P[((size_t)t_ * B_SZ + b_) * G_SZ + col] = acc[m][nt][r] + bv;
      }
    }
  }
}

// --------------------------------------------------------- LSTM recurrence
// grid = 2 (direction), block = 1024 thr = 32 waves, persistent over T steps.
// Wave w: hidden units j in [16w, 16w+16); gate columns {j, j+512, j+1024, j+1536}.
// Per step: gates = h @ Whh^T (WMMA, K=512) + x_proj[t]; cell update in regs;
// h fed back as bf16 through double-buffered LDS.
__global__ __launch_bounds__(1024) void lstm_recur_kernel(
    const float* __restrict__ Pf, const float* __restrict__ Pb,
    const unsigned short* __restrict__ Whf, const unsigned short* __restrict__ Whb,
    const int* __restrict__ lengths,
    float* __restrict__ Outf, float* __restrict__ Outrb) {
  extern __shared__ unsigned short smem_raw[];
  unsigned short* hb = smem_raw;                       // [2][64][512] bf16
  int* slen = (int*)(smem_raw + 2 * B_SZ * H_SZ);      // [64]

  const float* P           = blockIdx.x ? Pb : Pf;
  const unsigned short* Wh = blockIdx.x ? Whb : Whf;
  float* Out               = blockIdx.x ? Outrb : Outf;

  const int tid  = threadIdx.x;
  const int wave = tid >> 5;
  const int lane = tid & 31;
  const int lr   = lane & 15;
  const int hi   = lane >> 4;
  const int j0   = wave * 16;
  const int j    = j0 + lr;

  for (int i = tid; i < B_SZ * H_SZ; i += 1024) hb[i] = 0;   // h0 = 0
  if (tid < B_SZ) slen[tid] = lengths[tid];
  __syncthreads();

  float cst[4][8];
#pragma unroll
  for (int m = 0; m < 4; ++m)
#pragma unroll
    for (int r = 0; r < 8; ++r) cst[m][r] = 0.f;

  v8f zero = {0.f,0.f,0.f,0.f,0.f,0.f,0.f,0.f};
  int p = 0;
  for (int t = 0; t < T_SEQ; ++t) {
    const unsigned short* hcur = hb + p * (B_SZ * H_SZ);
    unsigned short* hnext      = hb + (p ^ 1) * (B_SZ * H_SZ);

    v8f acc[4][4];
#pragma unroll
    for (int m = 0; m < 4; ++m)
#pragma unroll
      for (int g = 0; g < 4; ++g) acc[m][g] = zero;

    for (int ks = 0; ks < H_SZ / 32; ++ks) {
      const int k0 = ks * 32;
      FragAB a[4];
#pragma unroll
      for (int m = 0; m < 4; ++m) {
        const unsigned short* ap = hcur + (m * 16 + lr) * H_SZ + k0 + hi * 8;
        a[m].q[0] = *(const uint4*)ap;        // ds_load_b128
        a[m].q[1] = *(const uint4*)(ap + 16);
      }
      FragAB bf[4];
#pragma unroll
      for (int g = 0; g < 4; ++g) {
        const unsigned short* bp =
            Wh + (size_t)(g * H_SZ + j0 + lr) * H_SZ + k0 + hi * 8;
        bf[g].q[0] = *(const uint4*)bp;
        bf[g].q[1] = *(const uint4*)(bp + 16);
        __builtin_prefetch((const void*)(bp + 32), 0, 1);  // next K slice
      }
#pragma unroll
      for (int m = 0; m < 4; ++m)
#pragma unroll
        for (int g = 0; g < 4; ++g)
          acc[m][g] = __builtin_amdgcn_wmma_f32_16x16x32_bf16(
              false, a[m].v, false, bf[g].v, (short)0, acc[m][g], false, false);
    }

    const float* Pt = P + (size_t)t * (B_SZ * G_SZ);
#pragma unroll
    for (int m = 0; m < 4; ++m) {
#pragma unroll
      for (int r = 0; r < 8; ++r) {
        const int b_ = m * 16 + r + hi * 8;
        const size_t pb = (size_t)b_ * G_SZ + j;
        const float gi = acc[m][0][r] + Pt[pb];
        const float gf = acc[m][1][r] + Pt[pb + H_SZ];
        const float gg = acc[m][2][r] + Pt[pb + 2 * H_SZ];
        const float go = acc[m][3][r] + Pt[pb + 3 * H_SZ];
        const float i_ = sigmoidf_(gi);
        const float f_ = sigmoidf_(gf);
        const float g_ = tanhf(gg);
        const float o_ = sigmoidf_(go);
        const float cn = f_ * cst[m][r] + i_ * g_;
        const float hn = o_ * tanhf(cn);
        const bool mk = t < slen[b_];                 // branchless masking
        cst[m][r] = mk ? cn : cst[m][r];
        Out[((size_t)b_ * T_SEQ + t) * H_SZ + j] = mk ? hn : 0.f;
        hnext[b_ * H_SZ + j] = mk ? f2bf(hn) : hcur[b_ * H_SZ + j];
      }
    }
    __syncthreads();   // writes to hnext visible; all reads of hcur complete
    p ^= 1;
  }
}

// ----------------------------------------------------------------- epilogue
__global__ __launch_bounds__(512) void softmax512_kernel(
    const float* __restrict__ w, float* __restrict__ s) {
  __shared__ float red[512];
  const int t = threadIdx.x;
  const float v = w[t];
  red[t] = v; __syncthreads();
  for (int off = 256; off > 0; off >>= 1) {
    if (t < off) red[t] = fmaxf(red[t], red[t + off]);
    __syncthreads();
  }
  const float mx = red[0]; __syncthreads();
  const float e = __expf(v - mx);
  red[t] = e; __syncthreads();
  for (int off = 256; off > 0; off >>= 1) {
    if (t < off) red[t] += red[t + off];
    __syncthreads();
  }
  s[t] = e / red[0];
}

// pooled[b][h2] ; forward uses scores[t], backward uses scores[len-1-t]
__global__ __launch_bounds__(128) void pool_kernel(
    const float* __restrict__ Outf, const float* __restrict__ Outrb,
    const float* __restrict__ scores, const int* __restrict__ lengths,
    float* __restrict__ pooled) {
  __shared__ float ssc[T_SEQ];
  const int b  = blockIdx.x;
  const int h2 = blockIdx.y * 128 + threadIdx.x;
  for (int i = threadIdx.x; i < T_SEQ; i += 128) ssc[i] = scores[i];
  __syncthreads();
  const int len = lengths[b];
  float acc = 0.f;
  if (h2 < H_SZ) {
    const float* base = Outf + (size_t)b * T_SEQ * H_SZ + h2;
    for (int t = 0; t < len; ++t) acc += base[(size_t)t * H_SZ] * ssc[t];
  } else {
    const float* base = Outrb + (size_t)b * T_SEQ * H_SZ + (h2 - H_SZ);
    for (int t = 0; t < len; ++t) acc += base[(size_t)t * H_SZ] * ssc[len - 1 - t];
  }
  pooled[b * OH_SZ + h2] = acc;
}

__global__ __launch_bounds__(256) void fc1_kernel(
    const float* __restrict__ pooled, const float* __restrict__ w,
    const float* __restrict__ bias, float* __restrict__ h1) {
  const int idx = blockIdx.x * 256 + threadIdx.x;       // 64*1024
  const int b = idx >> 10, o = idx & (OH_SZ - 1);
  const float4* xp = (const float4*)(pooled + (size_t)b * OH_SZ);
  const float4* wp = (const float4*)(w + (size_t)o * OH_SZ);
  float acc = bias[o];
  for (int k = 0; k < OH_SZ / 4; ++k) {
    const float4 xa = xp[k], wa = wp[k];
    acc += xa.x * wa.x + xa.y * wa.y + xa.z * wa.z + xa.w * wa.w;
  }
  h1[idx] = fmaxf(acc, 0.f);
}

__global__ __launch_bounds__(256) void fc2_kernel(
    const float* __restrict__ h1, const float* __restrict__ w,
    const float* __restrict__ bias, const int* __restrict__ lengths,
    float* __restrict__ out) {
  const int idx = blockIdx.x * 256 + threadIdx.x;       // 64*512
  const int b = idx >> 9, t = idx & (T_SEQ - 1);
  const float4* xp = (const float4*)(h1 + (size_t)b * OH_SZ);
  const float4* wp = (const float4*)(w + (size_t)t * OH_SZ);
  float acc = bias[t];
  for (int k = 0; k < OH_SZ / 4; ++k) {
    const float4 xa = xp[k], wa = wp[k];
    acc += xa.x * wa.x + xa.y * wa.y + xa.z * wa.z + xa.w * wa.w;
  }
  out[idx] = (t < lengths[b]) ? acc : -1e30f;
}

// ---------------------------------------------------------------------------
extern "C" void kernel_launch(void* const* d_in, const int* in_sizes, int n_in,
                              void* d_out, int out_size, void* d_ws, size_t ws_size,
                              hipStream_t stream) {
  (void)in_sizes; (void)n_in; (void)out_size; (void)ws_size;
  const float* x       = (const float*)d_in[0];
  const int*   lengths = (const int*)d_in[1];
  const float* attn_w  = (const float*)d_in[2];
  const float* Wih_f   = (const float*)d_in[3];
  const float* Whh_f   = (const float*)d_in[4];
  const float* bih_f   = (const float*)d_in[5];
  const float* bhh_f   = (const float*)d_in[6];
  const float* Wih_b   = (const float*)d_in[7];
  const float* Whh_b   = (const float*)d_in[8];
  const float* bih_b   = (const float*)d_in[9];
  const float* bhh_b   = (const float*)d_in[10];
  const float* fc1_w   = (const float*)d_in[11];
  const float* fc1_b   = (const float*)d_in[12];
  const float* fc2_w   = (const float*)d_in[13];
  const float* fc2_b   = (const float*)d_in[14];

  char* ws = (char*)d_ws;
  size_t off = 0;
  auto alloc = [&](size_t bytes) -> char* {
    char* p = ws + off;
    off = (off + bytes + 255) & ~(size_t)255;
    return p;
  };

  unsigned short* Xf    = (unsigned short*)alloc((size_t)B_SZ * T_SEQ * E_SZ * 2);
  unsigned short* Xb    = (unsigned short*)alloc((size_t)B_SZ * T_SEQ * E_SZ * 2);
  unsigned short* Wihf  = (unsigned short*)alloc((size_t)G_SZ * E_SZ * 2);
  unsigned short* Wihb  = (unsigned short*)alloc((size_t)G_SZ * E_SZ * 2);
  unsigned short* Whhf  = (unsigned short*)alloc((size_t)G_SZ * H_SZ * 2);
  unsigned short* Whhb  = (unsigned short*)alloc((size_t)G_SZ * H_SZ * 2);
  float* biasf  = (float*)alloc((size_t)G_SZ * 4);
  float* biasb  = (float*)alloc((size_t)G_SZ * 4);
  float* Pf     = (float*)alloc((size_t)T_SEQ * B_SZ * G_SZ * 4);   // 268 MB
  float* Pb     = (float*)alloc((size_t)T_SEQ * B_SZ * G_SZ * 4);   // 268 MB
  float* Outf   = (float*)alloc((size_t)B_SZ * T_SEQ * H_SZ * 4);   // 67 MB
  float* Outrb  = (float*)alloc((size_t)B_SZ * T_SEQ * H_SZ * 4);   // 67 MB
  float* scores = (float*)alloc((size_t)T_SEQ * 4);
  float* pooled = (float*)alloc((size_t)B_SZ * OH_SZ * 4);
  float* h1     = (float*)alloc((size_t)B_SZ * OH_SZ * 4);

  // ---- prep
  cvt_bf16_kernel<<<(G_SZ * E_SZ + 255) / 256, 256, 0, stream>>>(Wih_f, Wihf, G_SZ * E_SZ);
  cvt_bf16_kernel<<<(G_SZ * E_SZ + 255) / 256, 256, 0, stream>>>(Wih_b, Wihb, G_SZ * E_SZ);
  cvt_bf16_kernel<<<(G_SZ * H_SZ + 255) / 256, 256, 0, stream>>>(Whh_f, Whhf, G_SZ * H_SZ);
  cvt_bf16_kernel<<<(G_SZ * H_SZ + 255) / 256, 256, 0, stream>>>(Whh_b, Whhb, G_SZ * H_SZ);
  add_bias_kernel<<<(G_SZ + 255) / 256, 256, 0, stream>>>(bih_f, bhh_f, biasf, G_SZ);
  add_bias_kernel<<<(G_SZ + 255) / 256, 256, 0, stream>>>(bih_b, bhh_b, biasb, G_SZ);
  prep_x_kernel<<<(B_SZ * T_SEQ * E_SZ + 255) / 256, 256, 0, stream>>>(x, lengths, Xf, Xb);

  // ---- input projections (both directions), WMMA
  dim3 gp(B_SZ * T_SEQ / 64, G_SZ / 256, 2);
  xproj_gemm_kernel<<<gp, 256, 0, stream>>>(Xf, Xb, Wihf, Wihb, biasf, biasb, Pf, Pb);

  // ---- recurrence: 1 persistent workgroup per direction
  const size_t lds_bytes = (size_t)2 * B_SZ * H_SZ * 2 + B_SZ * 4;  // 128K + 256
  (void)hipFuncSetAttribute((const void*)lstm_recur_kernel,
                            hipFuncAttributeMaxDynamicSharedMemorySize,
                            (int)lds_bytes);
  lstm_recur_kernel<<<2, 1024, lds_bytes, stream>>>(Pf, Pb, Whhf, Whhb, lengths,
                                                    Outf, Outrb);

  // ---- attention pooling + FC head
  softmax512_kernel<<<1, 512, 0, stream>>>(attn_w, scores);
  pool_kernel<<<dim3(B_SZ, OH_SZ / 128), 128, 0, stream>>>(Outf, Outrb, scores,
                                                           lengths, pooled);
  fc1_kernel<<<B_SZ * OH_SZ / 256, 256, 0, stream>>>(pooled, fc1_w, fc1_b, h1);
  fc2_kernel<<<B_SZ * T_SEQ / 256, 256, 0, stream>>>(h1, fc2_w, fc2_b, lengths,
                                                     (float*)d_out);
}
